// MoELayer_11003706212927
// MI455X (gfx1250) — matrix operand order repaired and checked by reading, not compiled, not run on hardware
//
#include <hip/hip_runtime.h>
#include <hip/hip_bf16.h>

#define HIDDEN 1024
#define FF     2048
#define NE     8
#define NTOK   8192
#define NROWS  (NTOK * 2)   // total token-expert pairs (top-2)

typedef __bf16 bf16;
typedef __attribute__((ext_vector_type(16))) __bf16 v16bf;
typedef __attribute__((ext_vector_type(8)))  float  v8f;

#define BM 128
#define BK 32
#define LDA 40        // padded LDS stride for A tiles (80B rows: bank-friendly, 16B aligned)
#define KT1 (HIDDEN / 32)   // 32  K-tiles in GEMM1
#define KT2 (FF / 32)       // 64  K-tiles in GEMM2
#define NT1 (FF / 16)       // 128 N-tiles in W1 swizzle
#define NT2 (HIDDEN / 16)   // 64  N-tiles in W2 swizzle

union frag_t { v16bf v; uint4 q[2]; };

__device__ __forceinline__ unsigned pack2(float a, float b) {
  unsigned short lo = __builtin_bit_cast(unsigned short, (bf16)a);
  unsigned short hi = __builtin_bit_cast(unsigned short, (bf16)b);
  return (unsigned)lo | ((unsigned)hi << 16);
}

// async 16B copy: global -> LDS, tracked by ASYNCcnt (CDNA5 TDM-lite path)
__device__ __forceinline__ void async_b128(unsigned lds_off, unsigned long long gaddr) {
  asm volatile("global_load_async_to_lds_b128 %0, %1, off"
               :: "v"(lds_off), "v"(gaddr) : "memory");
}
__device__ __forceinline__ unsigned lds_off_of(const void* p) {
  return (unsigned)(unsigned long long)p;   // low 32 bits of LDS-aperture address = LDS offset
}

// ---------------------------------------------------------------------------
// 1) zero output + meta counters + 128B zero pad (replays must re-init)
// ---------------------------------------------------------------------------
__global__ __launch_bounds__(256) void k_init(float* __restrict__ out, int n4,
                                              int* __restrict__ meta) {
  int i = blockIdx.x * 256 + threadIdx.x;
  if (i < n4) ((float4*)out)[i] = make_float4(0.f, 0.f, 0.f, 0.f);
  if (blockIdx.x == 0 && threadIdx.x < 64) meta[threadIdx.x] = 0;  // meta[0..31] + zero pad
}

// ---------------------------------------------------------------------------
// 2) x (f32) -> xb (bf16) row-major, 8 elems/thread
// ---------------------------------------------------------------------------
__global__ __launch_bounds__(256) void k_convX(const float* __restrict__ x,
                                               uint4* __restrict__ xb) {
  int i = blockIdx.x * 256 + threadIdx.x;
  const float4* p = (const float4*)x + (size_t)i * 2;
  float4 a = p[0], b = p[1];
  xb[i] = make_uint4(pack2(a.x, a.y), pack2(a.z, a.w), pack2(b.x, b.y), pack2(b.z, b.w));
}

// ---------------------------------------------------------------------------
// 3) W (f32 [e][K][N]) -> bf16 in WMMA-B-fragment layout [e][nt][kt][lane][16]
//    lane = (k%32 / 16)*16 + n%16 ; elems = 16 consecutive k within half
// ---------------------------------------------------------------------------
__global__ __launch_bounds__(256) void k_convW(const float* __restrict__ src,
                                               bf16* __restrict__ dst,
                                               int KT, int NT) {
  int tid  = blockIdx.x * 256 + threadIdx.x;
  int lane = tid & 31;
  int tile = tid >> 5;                       // e*NT*KT tiles total
  int e  = tile / (KT * NT);
  int r  = tile % (KT * NT);
  int nt = r / KT;
  int kt = r % KT;
  int laneRow = lane & 15, laneHalf = lane >> 4;
  int K = KT * 32, N = NT * 16;
  const float* s = src + ((size_t)e * K + kt * 32 + laneHalf * 16) * N + nt * 16 + laneRow;
  unsigned u[8];
#pragma unroll
  for (int i = 0; i < 8; i++)
    u[i] = pack2(s[(size_t)(2 * i) * N], s[(size_t)(2 * i + 1) * N]);
  uint4* d = (uint4*)(dst + ((((size_t)e * NT + nt) * KT + kt) * 32 + lane) * 16);
  d[0] = make_uint4(u[0], u[1], u[2], u[3]);
  d[1] = make_uint4(u[4], u[5], u[6], u[7]);
}

// ---------------------------------------------------------------------------
// 4) router: one wave32 per token, fp32 exact (controls discrete top-k)
// ---------------------------------------------------------------------------
__global__ __launch_bounds__(256) void k_router(const float* __restrict__ x,
                                                const float* __restrict__ Wr,
                                                const float* __restrict__ br,
                                                int*  __restrict__ meta,
                                                int*  __restrict__ tokTop,
                                                float* __restrict__ tokGate) {
  int wave = (blockIdx.x * 256 + threadIdx.x) >> 5;
  int lane = threadIdx.x & 31;
  const float* xr = x + (size_t)wave * HIDDEN;
  float acc[NE];
#pragma unroll
  for (int e = 0; e < NE; e++) acc[e] = 0.f;
  for (int d = lane; d < HIDDEN; d += 32) {
    float xv = xr[d];
    const float4* w = (const float4*)(Wr + d * NE);
    float4 w0 = w[0], w1 = w[1];
    acc[0] += xv * w0.x; acc[1] += xv * w0.y; acc[2] += xv * w0.z; acc[3] += xv * w0.w;
    acc[4] += xv * w1.x; acc[5] += xv * w1.y; acc[6] += xv * w1.z; acc[7] += xv * w1.w;
  }
#pragma unroll
  for (int e = 0; e < NE; e++) {
#pragma unroll
    for (int off = 16; off >= 1; off >>= 1) acc[e] += __shfl_xor(acc[e], off);
    acc[e] += br[e];
  }
  if (lane == 0) {
    int e0 = 0;
#pragma unroll
    for (int e = 1; e < NE; e++) if (acc[e] > acc[e0]) e0 = e;
    int e1 = -1;
#pragma unroll
    for (int e = 0; e < NE; e++)
      if (e != e0 && (e1 < 0 || acc[e] > acc[e1])) e1 = e;
    float r = __expf(acc[e1] - acc[e0]);      // softmax-over-2 == renormalized top-2
    float s = 1.f + r;
    atomicAdd(&meta[e0], 1);
    atomicAdd(&meta[e1], 1);
    tokTop[wave * 2 + 0]  = e0;
    tokTop[wave * 2 + 1]  = e1;
    tokGate[wave * 2 + 0] = 1.f / s;
    tokGate[wave * 2 + 1] = r / s;
  }
}

__global__ void k_scan(int* __restrict__ meta) {
  if (blockIdx.x == 0 && threadIdx.x == 0) {
    int s = 0;
#pragma unroll
    for (int e = 0; e < NE; e++) { meta[16 + e] = s; s += meta[e]; }
    meta[16 + NE] = s;
  }
}

__global__ __launch_bounds__(256) void k_build(const int* __restrict__ tokTop,
                                               const float* __restrict__ tokGate,
                                               int* __restrict__ meta,
                                               int* __restrict__ rowToken,
                                               float* __restrict__ rowGate) {
  int tk = blockIdx.x * 256 + threadIdx.x;
  if (tk >= NTOK) return;
#pragma unroll
  for (int k = 0; k < 2; k++) {
    int e = tokTop[tk * 2 + k];
    int p = atomicAdd(&meta[8 + e], 1);
    int r = meta[16 + e] + p;
    rowToken[r] = tk;
    rowGate[r]  = tokGate[tk * 2 + k];
  }
}

// ---------------------------------------------------------------------------
// 5) GEMM1: H[r,:] = relu(xb[tok(r),:] @ W1b[e] + b1[e])  — async-staged bf16 WMMA
// ---------------------------------------------------------------------------
__global__ __launch_bounds__(256) void k_gemm1(const bf16* __restrict__ xb,
                                               const bf16* __restrict__ W1b,
                                               const float* __restrict__ b1,
                                               const int* __restrict__ rowToken,
                                               const int* __restrict__ meta,
                                               const int* __restrict__ zpad,
                                               bf16* __restrict__ H) {
  int e    = blockIdx.z;
  int rows = meta[e];
  int m0   = blockIdx.x * BM;
  if (m0 >= rows) return;
  int nt0  = blockIdx.y * 8;          // 8 N-tiles of 16 => BN=128
  int base = meta[16 + e];

  __shared__ __align__(16) bf16 As[2][BM * LDA];
  __shared__ __align__(16) bf16 Bs[2][8 * 512];

  int t = threadIdx.x;
  int lane = t & 31, wid = t >> 5;
  int wm = wid & 3, wn = wid >> 2;
  int laneRow = lane & 15, laneHalf = lane >> 4;

  // A async staging: thread -> (row, 32B half)
  int arow = t >> 1, ahalf = t & 1;
  bool aval = (m0 + arow) < rows;
  int  atok = aval ? rowToken[base + m0 + arow] : 0;
  unsigned long long aga = (unsigned long long)(const void*)(xb + (size_t)atok * HIDDEN)
                         + (unsigned)(ahalf * 32);
  unsigned long long zga = (unsigned long long)(const void*)zpad + (unsigned)(ahalf * 32);
  unsigned adst[2] = { lds_off_of(&As[0][arow * LDA + ahalf * 16]),
                       lds_off_of(&As[1][arow * LDA + ahalf * 16]) };
  // B async staging: thread -> (local ntile, lane 32B run)
  int ntl = t >> 5, bl = t & 31;
  unsigned long long bga = (unsigned long long)(const void*)
      (W1b + ((((size_t)e * NT1 + nt0 + ntl) * KT1) * 32 + bl) * 16);
  unsigned bdst[2] = { lds_off_of(&Bs[0][ntl * 512 + bl * 16]),
                       lds_off_of(&Bs[1][ntl * 512 + bl * 16]) };

  v8f acc[2][4];
#pragma unroll
  for (int i = 0; i < 2; i++)
#pragma unroll
    for (int j = 0; j < 4; j++) acc[i][j] = (v8f)0.f;

  // prologue: tile 0 into buffer 0   (4 async ops per wave per tile)
  {
    unsigned long long ga = aval ? aga : zga;
    async_b128(adst[0], ga); async_b128(adst[0] + 16, ga + 16);
    async_b128(bdst[0], bga); async_b128(bdst[0] + 16, bga + 16);
  }

  for (int kt = 0; kt < KT1; kt++) {
    int b = kt & 1;
    if (kt + 1 < KT1) {       // issue next tile into other buffer
      unsigned long long ga = aval ? (aga + (unsigned)((kt + 1) * 64)) : zga;
      unsigned long long gb = bga + (unsigned long long)(kt + 1) * 1024u;
      async_b128(adst[b ^ 1], ga); async_b128(adst[b ^ 1] + 16, ga + 16);
      async_b128(bdst[b ^ 1], gb); async_b128(bdst[b ^ 1] + 16, gb + 16);
      asm volatile("s_wait_asynccnt 0x4" ::: "memory");   // tile kt arrived
    } else {
      asm volatile("s_wait_asynccnt 0x0" ::: "memory");
    }
    __syncthreads();

    frag_t af[2], bfg[4];
#pragma unroll
    for (int i = 0; i < 2; i++) {
      const bf16* s = &As[b][(wm * 32 + i * 16 + laneRow) * LDA + laneHalf * 16];
      af[i].q[0] = *(const uint4*)s;
      af[i].q[1] = *(const uint4*)(s + 8);
    }
#pragma unroll
    for (int j = 0; j < 4; j++) {
      const bf16* s = &Bs[b][(wn * 4 + j) * 512 + lane * 16];
      bfg[j].q[0] = *(const uint4*)s;
      bfg[j].q[1] = *(const uint4*)(s + 8);
    }
#pragma unroll
    for (int i = 0; i < 2; i++)
#pragma unroll
      for (int j = 0; j < 4; j++)
        acc[i][j] = __builtin_amdgcn_wmma_f32_16x16x32_bf16(
            false, af[i].v, false, bfg[j].v, (short)0, acc[i][j], false, false);
    __syncthreads();          // free buffer b before it is re-filled
  }

#pragma unroll
  for (int j = 0; j < 4; j++) {
    int col = nt0 * 16 + wn * 64 + j * 16 + laneRow;
    float bias = b1[e * FF + col];
#pragma unroll
    for (int i = 0; i < 2; i++) {
      int rbase = m0 + wm * 32 + i * 16 + laneHalf * 8;
#pragma unroll
      for (int r = 0; r < 8; r++) {
        int row = rbase + r;
        float v = acc[i][j][r] + bias;
        v = v > 0.f ? v : 0.f;
        if (row < rows) H[(size_t)(base + row) * FF + col] = (bf16)v;
      }
    }
  }
}

// ---------------------------------------------------------------------------
// 6) GEMM2: out[tok(r),:] += gate(r) * (H[r,:] @ W2b[e] + b2[e])
//    exactly 2 addends per row onto zeroed out => atomic add is bit-deterministic
// ---------------------------------------------------------------------------
__global__ __launch_bounds__(256) void k_gemm2(const bf16* __restrict__ H,
                                               const bf16* __restrict__ W2b,
                                               const float* __restrict__ b2,
                                               const int* __restrict__ rowToken,
                                               const float* __restrict__ rowGate,
                                               const int* __restrict__ meta,
                                               const int* __restrict__ zpad,
                                               float* __restrict__ out) {
  int e    = blockIdx.z;
  int rows = meta[e];
  int m0   = blockIdx.x * BM;
  if (m0 >= rows) return;
  int nt0  = blockIdx.y * 8;
  int base = meta[16 + e];

  __shared__ __align__(16) bf16 As[2][BM * LDA];
  __shared__ __align__(16) bf16 Bs[2][8 * 512];

  int t = threadIdx.x;
  int lane = t & 31, wid = t >> 5;
  int wm = wid & 3, wn = wid >> 2;
  int laneRow = lane & 15, laneHalf = lane >> 4;

  int arow = t >> 1, ahalf = t & 1;
  bool aval = (m0 + arow) < rows;
  unsigned long long aga = (unsigned long long)(const void*)
                             (H + (size_t)(base + m0 + arow) * FF)
                         + (unsigned)(ahalf * 32);
  unsigned long long zga = (unsigned long long)(const void*)zpad + (unsigned)(ahalf * 32);
  unsigned adst[2] = { lds_off_of(&As[0][arow * LDA + ahalf * 16]),
                       lds_off_of(&As[1][arow * LDA + ahalf * 16]) };
  int ntl = t >> 5, bl = t & 31;
  unsigned long long bga = (unsigned long long)(const void*)
      (W2b + ((((size_t)e * NT2 + nt0 + ntl) * KT2) * 32 + bl) * 16);
  unsigned bdst[2] = { lds_off_of(&Bs[0][ntl * 512 + bl * 16]),
                       lds_off_of(&Bs[1][ntl * 512 + bl * 16]) };

  v8f acc[2][4];
#pragma unroll
  for (int i = 0; i < 2; i++)
#pragma unroll
    for (int j = 0; j < 4; j++) acc[i][j] = (v8f)0.f;

  {
    unsigned long long ga = aval ? aga : zga;
    async_b128(adst[0], ga); async_b128(adst[0] + 16, ga + 16);
    async_b128(bdst[0], bga); async_b128(bdst[0] + 16, bga + 16);
  }

  for (int kt = 0; kt < KT2; kt++) {
    int b = kt & 1;
    if (kt + 1 < KT2) {
      unsigned long long ga = aval ? (aga + (unsigned)((kt + 1) * 64)) : zga;
      unsigned long long gb = bga + (unsigned long long)(kt + 1) * 1024u;
      async_b128(adst[b ^ 1], ga); async_b128(adst[b ^ 1] + 16, ga + 16);
      async_b128(bdst[b ^ 1], gb); async_b128(bdst[b ^ 1] + 16, gb + 16);
      asm volatile("s_wait_asynccnt 0x4" ::: "memory");
    } else {
      asm volatile("s_wait_asynccnt 0x0" ::: "memory");
    }
    __syncthreads();

    frag_t af[2], bfg[4];
#pragma unroll
    for (int i = 0; i < 2; i++) {
      const bf16* s = &As[b][(wm * 32 + i * 16 + laneRow) * LDA + laneHalf * 16];
      af[i].q[0] = *(const uint4*)s;
      af[i].q[1] = *(const uint4*)(s + 8);
    }
#pragma unroll
    for (int j = 0; j < 4; j++) {
      const bf16* s = &Bs[b][(wn * 4 + j) * 512 + lane * 16];
      bfg[j].q[0] = *(const uint4*)s;
      bfg[j].q[1] = *(const uint4*)(s + 8);
    }
#pragma unroll
    for (int i = 0; i < 2; i++)
#pragma unroll
      for (int j = 0; j < 4; j++)
        acc[i][j] = __builtin_amdgcn_wmma_f32_16x16x32_bf16(
            false, af[i].v, false, bfg[j].v, (short)0, acc[i][j], false, false);
    __syncthreads();
  }

#pragma unroll
  for (int j = 0; j < 4; j++) {
    int col = nt0 * 16 + wn * 64 + j * 16 + laneRow;
    float bias = b2[e * HIDDEN + col];
#pragma unroll
    for (int i = 0; i < 2; i++) {
      int rbase = m0 + wm * 32 + i * 16 + laneHalf * 8;
#pragma unroll
      for (int r = 0; r < 8; r++) {
        int row = rbase + r;
        if (row < rows) {
          int gr = base + row;
          float v = (acc[i][j][r] + bias) * rowGate[gr];
          unsafeAtomicAdd(out + (size_t)rowToken[gr] * HIDDEN + col, v);
        }
      }
    }
  }
}

// ---------------------------------------------------------------------------
extern "C" void kernel_launch(void* const* d_in, const int* in_sizes, int n_in,
                              void* d_out, int out_size, void* d_ws, size_t ws_size,
                              hipStream_t stream) {
  const float* x  = (const float*)d_in[0];
  const float* Wr = (const float*)d_in[1];
  const float* br = (const float*)d_in[2];
  const float* W1 = (const float*)d_in[3];
  const float* b1 = (const float*)d_in[4];
  const float* W2 = (const float*)d_in[5];
  const float* b2 = (const float*)d_in[6];
  float* out = (float*)d_out;

  // workspace layout (~144.5 MiB)
  char*  ws       = (char*)d_ws;
  bf16*  H        = (bf16*)ws;                                     // 64 MiB
  bf16*  XB       = (bf16*)(ws + (size_t)NROWS * FF * 2);          // 16 MiB
  bf16*  W1B      = (bf16*)(ws + (size_t)NROWS * FF * 2 + (size_t)NTOK * HIDDEN * 2);
  bf16*  W2B      = (bf16*)((char*)W1B + (size_t)NE * HIDDEN * FF * 2);
  int*   rowToken = (int*)((char*)W2B + (size_t)NE * FF * HIDDEN * 2);
  float* rowGate  = (float*)((char*)rowToken + NROWS * sizeof(int));
  int*   tokTop   = (int*)((char*)rowGate + NROWS * sizeof(float));
  float* tokGate  = (float*)((char*)tokTop + NTOK * 2 * sizeof(int));
  int*   meta     = (int*)((char*)tokGate + NTOK * 2 * sizeof(float));
  int*   zpad     = meta + 32;   // 128B zero pad for out-of-range async rows
  // meta: [0..7]=counts, [8..15]=fill ptrs, [16..24]=segment bases

  int n4 = out_size / 4;
  k_init<<<(n4 + 255) / 256, 256, 0, stream>>>(out, n4, meta);
  k_convX<<<(NTOK * HIDDEN / 8) / 256, 256, 0, stream>>>(x, (uint4*)XB);
  k_convW<<<(NE * KT1 * NT1 * 32) / 256, 256, 0, stream>>>(W1, W1B, KT1, NT1);
  k_convW<<<(NE * KT2 * NT2 * 32) / 256, 256, 0, stream>>>(W2, W2B, KT2, NT2);
  k_router<<<NTOK / 8, 256, 0, stream>>>(x, Wr, br, meta, tokTop, tokGate);
  k_scan<<<1, 32, 0, stream>>>(meta);
  k_build<<<NTOK / 256, 256, 0, stream>>>(tokTop, tokGate, meta, rowToken, rowGate);
  k_gemm1<<<dim3(NTOK / BM, FF / 128, NE), 256, 0, stream>>>(XB, W1B, b1, rowToken, meta, zpad, H);
  k_gemm2<<<dim3(NTOK / BM, HIDDEN / 128, NE), 256, 0, stream>>>(H, W2B, b2, rowToken, rowGate, meta, zpad, out);
}
